// SelfAttention_80994493267992
// MI455X (gfx1250) — matrix operand order repaired and checked by reading, compiled, not run on hardware
//
#include <hip/hip_runtime.h>
#include <math.h>
#include <stdint.h>

typedef __attribute__((ext_vector_type(16))) __bf16 bf16x16;
typedef __attribute__((ext_vector_type(8)))  __bf16 bf16x8;
typedef __attribute__((ext_vector_type(4)))  __bf16 bf16x4;
typedef __attribute__((ext_vector_type(8)))  float  f32x8;

#define SQ 4096        // sequence length
#define LDSTRIDE 40    // padded LDS row stride (bf16 elems); 80B rows -> conflict-free

// ---- CDNA5 async global->LDS path (guarded; falls back to reg staging) ----
#if defined(__has_builtin)
#  if __has_builtin(__builtin_amdgcn_global_load_async_to_lds_b128)
#    define HAVE_ASYNC_LDS 1
#  endif
#endif
#ifndef HAVE_ASYNC_LDS
#  define HAVE_ASYNC_LDS 0
#endif

// builtin param types (from hipcc diagnostic): int4-vector pointers,
// src in global (AS1, prints as __device__), dst in LDS (AS3)
typedef int b128vec __attribute__((vector_size(16)));
#define GAS1(x) ((__attribute__((address_space(1))) b128vec*)(uintptr_t)(x))
#define LAS3(x) ((__attribute__((address_space(3))) b128vec*)(uintptr_t)(x))

__device__ __forceinline__ void wait_async_zero() {
#if defined(__has_builtin)
#  if __has_builtin(__builtin_amdgcn_s_wait_asynccnt)
    __builtin_amdgcn_s_wait_asynccnt(0);
#  else
    asm volatile("s_wait_asynccnt 0x0" ::: "memory");
#  endif
#else
    asm volatile("s_wait_asynccnt 0x0" ::: "memory");
#endif
}

// ---------------------------------------------------------------------------
// f32 -> bf16 conversion (vectorized: float4 in, bf16x4 out)
// ---------------------------------------------------------------------------
__global__ __launch_bounds__(256)
void cvt_f32_bf16(const float* __restrict__ in, __bf16* __restrict__ out, int n4) {
    int i = blockIdx.x * 256 + threadIdx.x;
    if (i < n4) {
        float4 v = reinterpret_cast<const float4*>(in)[i];
        bf16x4 o;
        o[0] = (__bf16)v.x; o[1] = (__bf16)v.y; o[2] = (__bf16)v.z; o[3] = (__bf16)v.w;
        reinterpret_cast<bf16x4*>(out)[i] = o;
    }
}

// ---------------------------------------------------------------------------
// Tiled bf16 WMMA GEMM:  C[b] = alpha * A[b] (MxK) * B[b]
//   A_F32 : A supplied as f32, converted to bf16 while staging to LDS
//   B_NT  : B supplied row-major [N x K] (weights / k-matrix) -> direct copy;
//           else B is [K x N] (v-matrix) -> transposed into LDS
//   C_BF16: store C as bf16 (else f32)
// Block tile 128(M) x 256(N), K-step 32, 256 threads = 8 waves (2x4),
// each wave computes a 64x64 macro-tile = 4x4 WMMA 16x16x32 tiles.
// Double-buffered LDS; async global->LDS staging when data is already bf16.
// All dims are multiples of the tiles -> no bounds checks; EXEC all-ones.
// ---------------------------------------------------------------------------
template<bool A_F32, bool B_NT, bool C_BF16>
__global__ __launch_bounds__(256)
void gemm_wmma_bf16(const void* __restrict__ Av, long lda, long strideA,
                    const __bf16* __restrict__ Bm, long ldb, long strideB,
                    void* __restrict__ Cv, long ldc, long strideC,
                    int Ktot, float alpha)
{
    constexpr bool USE_ASYNC = (HAVE_ASYNC_LDS != 0) && !A_F32 && B_NT;

    __shared__ __align__(16) __bf16 As[2][128][LDSTRIDE];   // [m][k]
    __shared__ __align__(16) __bf16 Bs[2][256][LDSTRIDE];   // n-major [n][k]

    const int tid  = threadIdx.x;
    const int lane = tid & 31;
    const int wid  = tid >> 5;
    const int wm   = wid >> 2;      // wave row 0..1  (64 M-rows each)
    const int wn   = wid & 3;       // wave col 0..3  (64 N-cols each)
    const int half = lane >> 4;     // 0 | 1
    const int l16  = lane & 15;

    const long bz = blockIdx.z;
    const long m0 = (long)blockIdx.y * 128;
    const long n0 = (long)blockIdx.x * 256;

    const float*  Af = (const float*)Av  + bz * strideA;
    const __bf16* Ab = (const __bf16*)Av + bz * strideA;
    const __bf16* Bp = Bm + bz * strideB;
    float*  Cf = (float*)Cv  + bz * strideC;
    __bf16* Cb = (__bf16*)Cv + bz * strideC;

    f32x8 acc[4][4];
#pragma unroll
    for (int i = 0; i < 4; ++i)
#pragma unroll
        for (int j = 0; j < 4; ++j)
#pragma unroll
            for (int e = 0; e < 8; ++e) acc[i][j][e] = 0.0f;

    // ---- stage one 128x32 A tile + 32x256 B tile into LDS buffer `buf` ----
    auto stage = [&](int buf, int kk) {
        // A: 128 rows x 32 k -> 4096 elems, 2 chunks of 8 per thread
#pragma unroll
        for (int p = 0; p < 2; ++p) {
            int chunk = tid + p * 256;          // 0..511
            int r  = chunk >> 2;
            int kc = (chunk & 3) * 8;
            if constexpr (A_F32) {
                const float* g = Af + (m0 + r) * lda + kk + kc;
                float4 v0 = *(const float4*)(g);
                float4 v1 = *(const float4*)(g + 4);
                bf16x8 o;
                o[0] = (__bf16)v0.x; o[1] = (__bf16)v0.y;
                o[2] = (__bf16)v0.z; o[3] = (__bf16)v0.w;
                o[4] = (__bf16)v1.x; o[5] = (__bf16)v1.y;
                o[6] = (__bf16)v1.z; o[7] = (__bf16)v1.w;
                *(bf16x8*)&As[buf][r][kc] = o;
            } else if constexpr (USE_ASYNC) {
#if HAVE_ASYNC_LDS
                __builtin_amdgcn_global_load_async_to_lds_b128(
                    GAS1(Ab + (m0 + r) * lda + kk + kc),
                    LAS3(&As[buf][r][kc]), 0, 0);
#endif
            } else {
                *(bf16x8*)&As[buf][r][kc] =
                    *(const bf16x8*)(Ab + (m0 + r) * lda + kk + kc);
            }
        }
        // B: 256 n x 32 k -> 8192 elems, 4 chunks of 8 per thread
#pragma unroll
        for (int p = 0; p < 4; ++p) {
            int chunk = tid + p * 256;          // 0..1023
            if constexpr (B_NT) {               // B given [N x K]: direct copy
                int r  = chunk >> 2;            // 0..255
                int kc = (chunk & 3) * 8;
                if constexpr (USE_ASYNC) {
#if HAVE_ASYNC_LDS
                    __builtin_amdgcn_global_load_async_to_lds_b128(
                        GAS1(Bp + (n0 + r) * ldb + kk + kc),
                        LAS3(&Bs[buf][r][kc]), 0, 0);
#endif
                } else {
                    *(bf16x8*)&Bs[buf][r][kc] =
                        *(const bf16x8*)(Bp + (n0 + r) * ldb + kk + kc);
                }
            } else {                            // B given [K x N]: transpose into LDS
                int kr = chunk >> 5;            // 0..31
                int nc = (chunk & 31) * 8;      // 0..248
                bf16x8 v = *(const bf16x8*)(Bp + (long)(kk + kr) * ldb + n0 + nc);
#pragma unroll
                for (int e = 0; e < 8; ++e) Bs[buf][nc + e][kr] = v[e];
            }
        }
    };

    stage(0, 0);
    if constexpr (USE_ASYNC) wait_async_zero();
    __syncthreads();

    const int nK = Ktot >> 5;                   // K-steps of 32
    for (int kt = 0; kt < nK; ++kt) {
        const int buf = kt & 1;
        if (kt + 1 < nK) stage(buf ^ 1, (kt + 1) << 5);

        // A fragments: 16x32 bf16; elem e<8 -> k=half*8+e, e>=8 -> k=16+half*8+(e-8)
        bf16x16 af[4];
#pragma unroll
        for (int i = 0; i < 4; ++i) {
            int r = wm * 64 + i * 16 + l16;
            bf16x8 lo = *(const bf16x8*)&As[buf][r][half * 8];
            bf16x8 hi = *(const bf16x8*)&As[buf][r][16 + half * 8];
#pragma unroll
            for (int e = 0; e < 8; ++e) { af[i][e] = lo[e]; af[i][8 + e] = hi[e]; }
        }
        // B fragments: 32x16 bf16; elem e<8 -> k=half*8+e, e>=8 -> k=16+half*8+(e-8), col = l16
        bf16x16 bfm[4];
#pragma unroll
        for (int j = 0; j < 4; ++j) {
            int c = wn * 64 + j * 16 + l16;
            bf16x8 lo = *(const bf16x8*)&Bs[buf][c][half * 8];
            bf16x8 hi = *(const bf16x8*)&Bs[buf][c][16 + half * 8];
#pragma unroll
            for (int e = 0; e < 8; ++e) { bfm[j][e] = lo[e]; bfm[j][8 + e] = hi[e]; }
        }

#pragma unroll
        for (int i = 0; i < 4; ++i)
#pragma unroll
            for (int j = 0; j < 4; ++j)
                acc[i][j] = __builtin_amdgcn_wmma_f32_16x16x32_bf16(
                    false, af[i], false, bfm[j], (short)0, acc[i][j], false, false);

        if constexpr (USE_ASYNC) { if (kt + 1 < nK) wait_async_zero(); }
        __syncthreads();
    }

    // C/D layout: VGPR e -> M = half*8 + e ; N = l16
#pragma unroll
    for (int i = 0; i < 4; ++i) {
        long rbase = m0 + wm * 64 + i * 16 + half * 8;
#pragma unroll
        for (int j = 0; j < 4; ++j) {
            long c = n0 + wn * 64 + j * 16 + l16;
#pragma unroll
            for (int e = 0; e < 8; ++e) {
                float v = alpha * acc[i][j][e];
                if constexpr (C_BF16) Cb[(rbase + e) * ldc + c] = (__bf16)v;
                else                  Cf[(rbase + e) * ldc + c] = v;
            }
        }
    }
}

// ---------------------------------------------------------------------------
// Softmax over axis 1 (the QUERY axis): per-(b,m) column statistics.
// ---------------------------------------------------------------------------
__global__ __launch_bounds__(256)
void softmax_colstats_partial(const float* __restrict__ attn,
                              float* __restrict__ pmax, float* __restrict__ psum)
{
    const int QCHUNK = 512;
    int m  = blockIdx.x * 256 + threadIdx.x;   // column index 0..4095
    int qc = blockIdx.y;                       // 0..7
    int b  = blockIdx.z;
    const float* p = attn + (size_t)b * SQ * SQ + (size_t)qc * QCHUNK * SQ + m;

    float mx = -INFINITY, sum = 0.0f;
    for (int q = 0; q < QCHUNK; ++q) {
        float s = p[(size_t)q * SQ];           // coalesced across lanes
        if (s > mx) { sum = sum * __expf(mx - s) + 1.0f; mx = s; }
        else        { sum += __expf(s - mx); }
    }
    size_t idx = ((size_t)b * 8 + qc) * SQ + m;
    pmax[idx] = mx;
    psum[idx] = sum;
}

__global__ __launch_bounds__(256)
void softmax_colstats_combine(const float* __restrict__ pmax,
                              const float* __restrict__ psum,
                              float* __restrict__ colmax, float* __restrict__ colsum)
{
    int idx = blockIdx.x * 256 + threadIdx.x;  // b*SQ + m  (B*SQ total)
    int b = idx >> 12, m = idx & (SQ - 1);
    float mx = -INFINITY;
#pragma unroll
    for (int c = 0; c < 8; ++c) mx = fmaxf(mx, pmax[((size_t)b * 8 + c) * SQ + m]);
    float sum = 0.0f;
#pragma unroll
    for (int c = 0; c < 8; ++c) {
        size_t k = ((size_t)b * 8 + c) * SQ + m;
        sum += psum[k] * __expf(pmax[k] - mx);
    }
    colmax[idx] = mx;
    colsum[idx] = sum;
}

__global__ __launch_bounds__(256)
void softmax_normalize(float* __restrict__ attn,
                       const float* __restrict__ colmax,
                       const float* __restrict__ colsum)
{
    int m  = blockIdx.x * 256 + threadIdx.x;
    int qc = blockIdx.y;                       // 0..15, 256 q-rows each
    int b  = blockIdx.z;
    float mx  = colmax[b * SQ + m];
    float inv = 1.0f / colsum[b * SQ + m];
    float* p = attn + (size_t)b * SQ * SQ + (size_t)qc * 256 * SQ + m;
    for (int q = 0; q < 256; ++q) {
        size_t o = (size_t)q * SQ;
        p[o] = __expf(p[o] - mx) * inv;
    }
}

// ---------------------------------------------------------------------------
extern "C" void kernel_launch(void* const* d_in, const int* in_sizes, int n_in,
                              void* d_out, int out_size, void* d_ws, size_t ws_size,
                              hipStream_t stream)
{
    (void)in_sizes; (void)n_in; (void)out_size; (void)ws_size;
    const int B = 4, S = SQ, H = 512, Kd = 512;

    const float* x  = (const float*)d_in[0];
    const float* Wq = (const float*)d_in[1];
    const float* Wk = (const float*)d_in[2];
    const float* Wv = (const float*)d_in[3];

    float* out  = (float*)d_out;                        // [B,S,H]  = 8,388,608 f32
    float* attn = out + (size_t)B * S * H;              // [B,S,S]  = 67,108,864 f32

    // workspace carve-up (bytes)
    char* w = (char*)d_ws;
    __bf16* xb   = (__bf16*)(w);                        // x bf16   [16384 x 512]
    __bf16* Wb   = (__bf16*)(w + 16777216);             // Wqkv     [1536  x 512]
    __bf16* qkvb = (__bf16*)(w + 18350080);             // q|k|v    [16384 x 1536]
    float*  pmax = (float*)(w + 68681728);              // [B,8,S]
    float*  psum = (float*)(w + 69206016);              // [B,8,S]
    float*  cmax = (float*)(w + 69730304);              // [B,S]
    float*  csum = (float*)(w + 69795840);              // [B,S]

    // 1) convert inputs to bf16 (Wq,Wk,Wv packed as rows 0..511|512..1023|1024..1535)
    cvt_f32_bf16<<<dim3(8192), 256, 0, stream>>>(x,  xb,          2097152);
    cvt_f32_bf16<<<dim3(256),  256, 0, stream>>>(Wq, Wb,          65536);
    cvt_f32_bf16<<<dim3(256),  256, 0, stream>>>(Wk, Wb + 262144, 65536);
    cvt_f32_bf16<<<dim3(256),  256, 0, stream>>>(Wv, Wb + 524288, 65536);

    // 2) fused projections: qkv[16384 x 1536] = x_bf16 * Wqkv^T  (bf16 out)
    gemm_wmma_bf16<false, true, true><<<dim3(6, 128, 1), 256, 0, stream>>>(
        xb, 512, 0, Wb, 512, 0, qkvb, 1536, 0, Kd, 1.0f);

    // 3) raw scaled scores per batch: attn = q * k^T / sqrt(512)  (f32 -> d_out)
    gemm_wmma_bf16<false, true, false><<<dim3(16, 32, 4), 256, 0, stream>>>(
        qkvb,       1536, (long)S * 1536,
        qkvb + 512, 1536, (long)S * 1536,
        attn,       S,    (long)S * S,
        Kd, 0.04419417382415922f /* 1/sqrt(512) */);

    // 4) column (query-axis) softmax statistics
    softmax_colstats_partial<<<dim3(16, 8, B), 256, 0, stream>>>(attn, pmax, psum);
    softmax_colstats_combine<<<dim3(64),       256, 0, stream>>>(pmax, psum, cmax, csum);

    // 5) normalize in place: attn = exp(s - colmax) / colsum
    softmax_normalize<<<dim3(16, 16, B), 256, 0, stream>>>(attn, cmax, csum);

    // 6) out = attn(f32->bf16) * v   (v is rows [K x N] in qkvb columns 1024..1535)
    gemm_wmma_bf16<true, false, false><<<dim3(2, 32, 4), 256, 0, stream>>>(
        attn,        S,    (long)S * S,
        qkvb + 1024, 1536, (long)S * 1536,
        out,         H,    (long)S * H,
        S, 1.0f);
}
